// MultiQueryAttention_78872779424175
// MI455X (gfx1250) — compile-verified, matrix operand order checked by reading, not run on hardware
//
#include <hip/hip_runtime.h>

// ---------------------------------------------------------------------------
// MQA forward for MI455X (gfx1250, wave32, WMMA + TDM).
//   cvt f32->bf16 (Q, K, W_q, W_kv, W_o)
//   q  = Qb @ W_qb^T + b_q    (bf16 WMMA GEMM)
//   kv = Kb @ W_kvb^T + b_kv  (bf16 WMMA GEMM)
//   attn: double-buffered TDM-staged 64-key K/V tiles, cooperative V
//         transpose in LDS, WMMA scores + exp2-domain online softmax +
//         WMMA context, ctx emitted bf16
//   out = ctx @ W_ob^T + b_o  (bf16 WMMA GEMM, f32 out)
// Workspace: ~39 MB bf16 staging.
// ---------------------------------------------------------------------------

#define DEV __device__ __forceinline__

typedef __attribute__((ext_vector_type(16))) __bf16       v16bf;
typedef __attribute__((ext_vector_type(8)))  float        v8f;
typedef __attribute__((ext_vector_type(4)))  unsigned int u32x4;
typedef __attribute__((ext_vector_type(8)))  int          i32x8;
typedef __attribute__((ext_vector_type(4)))  int          i32x4;

static constexpr int D_MODEL = 1024;
static constexpr int D_K     = 64;
static constexpr int SEQ     = 2048;
static constexpr int ROWS    = 4096;   // B * S
static constexpr float SC2   = 0.125f * 1.44269504088896340736f; // scale*log2e

DEV unsigned short f2bfbits(float f) {
    union { float f; unsigned u; } v; v.f = f;
    unsigned r = v.u + 0x7FFFu + ((v.u >> 16) & 1u);
    return (unsigned short)(r >> 16);
}
DEV __bf16 f2bf(float f) {
    unsigned short h = f2bfbits(f);
    __bf16 b; __builtin_memcpy(&b, &h, 2);
    return b;
}

DEV v8f wmma_bf16(v16bf a, v16bf b, v8f c) {
    return __builtin_amdgcn_wmma_f32_16x16x32_bf16(
        false, a, false, b, (short)0, c, false, false);
}

DEV v16bf ldfrag(const __bf16* p0, const __bf16* p1) {
    v16bf r;
#pragma unroll
    for (int j = 0; j < 8; ++j) { r[j] = p0[j]; r[j + 8] = p1[j]; }
    return r;
}

// ---------------------------------------------------------------------------
__global__ __launch_bounds__(256)
void cvt_bf16(const float* __restrict__ s, unsigned short* __restrict__ d, int n8) {
    const int i = blockIdx.x * 256 + threadIdx.x;
    if (i >= n8) return;
    const float4 a = ((const float4*)s)[2 * i];
    const float4 b = ((const float4*)s)[2 * i + 1];
    uint4 o;
    o.x = ((unsigned)f2bfbits(a.y) << 16) | f2bfbits(a.x);
    o.y = ((unsigned)f2bfbits(a.w) << 16) | f2bfbits(a.z);
    o.z = ((unsigned)f2bfbits(b.y) << 16) | f2bfbits(b.x);
    o.w = ((unsigned)f2bfbits(b.w) << 16) | f2bfbits(b.z);
    ((uint4*)d)[i] = o;
}

// ---------------------------------------------------------------------------
// C(MxN) = A(MxK,bf16) * W(NxK,bf16)^T + bias(N).  4 waves, 32x64 per wave.
// ---------------------------------------------------------------------------
template <bool OUT_BF16>
__global__ __launch_bounds__(128)
void gemm_bf16_wmma(const unsigned short* __restrict__ Au,
                    const unsigned short* __restrict__ Wu,
                    const float* __restrict__ bias, void* __restrict__ Cout,
                    int M, int N, int K) {
    const __bf16* A = (const __bf16*)Au;
    const __bf16* W = (const __bf16*)Wu;
    const int lane = threadIdx.x & 31;
    const int wave = threadIdx.x >> 5;
    const int row0 = blockIdx.y * 64 + (wave >> 1) * 32;
    const int col0 = blockIdx.x * 128 + (wave & 1) * 64;
    const int mn   = lane & 15;
    const int half = lane >> 4;

    const v8f vzero = {0.f, 0.f, 0.f, 0.f, 0.f, 0.f, 0.f, 0.f};
    v8f acc[2][4];
#pragma unroll
    for (int i = 0; i < 2; ++i)
#pragma unroll
        for (int j = 0; j < 4; ++j) acc[i][j] = vzero;

    const __bf16* ar0 = A + (size_t)(row0 + mn)      * K;
    const __bf16* ar1 = A + (size_t)(row0 + 16 + mn) * K;
    const __bf16* wr[4];
#pragma unroll
    for (int j = 0; j < 4; ++j) wr[j] = W + (size_t)(col0 + j * 16 + mn) * K;

    for (int kb = 0; kb < K; kb += 32) {
        const int k0 = kb + half * 8;
        const int k1 = k0 + 16;
        const v16bf a0 = ldfrag(ar0 + k0, ar0 + k1);
        const v16bf a1 = ldfrag(ar1 + k0, ar1 + k1);
#pragma unroll
        for (int j = 0; j < 4; ++j) {
            const v16bf bf = ldfrag(wr[j] + k0, wr[j] + k1);
            acc[0][j] = wmma_bf16(a0, bf, acc[0][j]);
            acc[1][j] = wmma_bf16(a1, bf, acc[1][j]);
        }
    }

    float bc[4];
#pragma unroll
    for (int j = 0; j < 4; ++j) bc[j] = bias[col0 + j * 16 + mn];
#pragma unroll
    for (int i = 0; i < 2; ++i)
#pragma unroll
        for (int r = 0; r < 8; ++r) {
            const int row = row0 + i * 16 + r + 8 * half;
#pragma unroll
            for (int j = 0; j < 4; ++j) {
                const size_t idx = (size_t)row * N + col0 + j * 16 + mn;
                const float v = acc[i][j][r] + bc[j];
                if (OUT_BF16) ((unsigned short*)Cout)[idx] = f2bfbits(v);
                else          ((float*)Cout)[idx] = v;
            }
        }
}

// ---------------------------------------------------------------------------
// Flash attention, 64-key steps, double-buffered TDM.
// Block = 4 waves sharing (b, h); wave owns a 16-row q tile.
// ---------------------------------------------------------------------------
__global__ __launch_bounds__(128)
void attn_kernel(const unsigned short* __restrict__ qws,
                 const unsigned short* __restrict__ kvws,
                 unsigned short* __restrict__ ctx) {
    __shared__ unsigned short KV[2][64 * 128];   // 2 x 16 KB K/V tiles
    __shared__ __bf16 VT[64][64];                // 8 KB transposed V tile
    __shared__ __bf16 Plds[4][16][64];           // 8 KB P (A-layout staging)

    const int tid  = threadIdx.x;
    const int lane = tid & 31;
    const int wave = tid >> 5;
    const int gw   = blockIdx.x * 4 + wave;
    const int qt   = gw & 127;            // block: same (b, h)
    const int h    = (gw >> 7) & 15;
    const int b    = gw >> 11;
    const int mn   = lane & 15;
    const int half = lane >> 4;

    const __bf16* q = (const __bf16*)qws;
    const unsigned short* kvb = kvws + (size_t)b * SEQ * 128;

    // Q fragments (A-layout), d split 0..31 / 32..63.
    const __bf16* qrow =
        q + ((size_t)(b * SEQ + qt * 16 + mn)) * D_MODEL + h * D_K;
    const v16bf qa0 = ldfrag(qrow + half * 8,      qrow + 16 + half * 8);
    const v16bf qa1 = ldfrag(qrow + 32 + half * 8, qrow + 48 + half * 8);

    const v8f vzero = {0.f, 0.f, 0.f, 0.f, 0.f, 0.f, 0.f, 0.f};
    v8f acc[4] = {vzero, vzero, vzero, vzero};
    float mrow[8], lrow[8];
#pragma unroll
    for (int r = 0; r < 8; ++r) { mrow[r] = -1e30f; lrow[r] = 0.f; }

    const unsigned lds_kv[2] = {
        (unsigned)(unsigned long long)(size_t)(void*)&KV[0][0],
        (unsigned)(unsigned long long)(size_t)(void*)&KV[1][0]};
    const i32x4 zero4 = {0, 0, 0, 0};
    const i32x8 zero8 = {0, 0, 0, 0, 0, 0, 0, 0};

    // D# for a 64x128 bf16 tile at row j0 of this batch's kv, into buffer bi.
    auto tdm_issue = [&](int j0, int bi) {
        const unsigned long long ga =
            (unsigned long long)(size_t)(kvb + (size_t)j0 * 128);
        u32x4 g0;
        g0[0] = 1u;                                   // count = 1 valid D#
        g0[1] = lds_kv[bi];                           // lds_addr
        g0[2] = (unsigned)ga;                         // global_addr[31:0]
        g0[3] = ((unsigned)(ga >> 32) & 0x01FFFFFFu) | (2u << 30); // type=2
        i32x8 g1;
        g1[0] = 0x00010000;          // workgroup_mask=0, data_size=1 (2B)
        g1[1] = (int)(128u << 16);   // tensor_dim0 = 128
        g1[2] = (int)(64u  << 16);   // tensor_dim1 = 64
        g1[3] = (int)(128u << 16);   // tile_dim0 = 128
        g1[4] = 64;                  // tile_dim1 = 64, tile_dim2 = 0
        g1[5] = 128;                 // tensor_dim0_stride = 128
        g1[6] = 0;
        g1[7] = 0;
        __builtin_amdgcn_tensor_load_to_lds(g0, g1, zero4, zero4, zero8, 0);
    };

    if (wave == 0) tdm_issue(0, 0);           // prime the pipeline

    for (int j0 = 0; j0 < SEQ; j0 += 64) {
        const int cur = (j0 >> 6) & 1;
        if (wave == 0) {
            if (j0 + 64 < SEQ) {
                tdm_issue(j0 + 64, cur ^ 1);              // overlap next DMA
                __builtin_amdgcn_s_wait_tensorcnt(1);     // current tile done
            } else {
                __builtin_amdgcn_s_wait_tensorcnt(0);
            }
        }
        if (j0 + 128 < SEQ)                                // warm L2 further out
            __builtin_prefetch((const char*)(kvb + (size_t)(j0 + 128) * 128) +
                                   tid * 128, 0, 1);
        __syncthreads();                                   // tile[cur] ready

        const __bf16* tile = (const __bf16*)KV[cur];

        // ---- cooperative V transpose: VT[d][key] = tile[key][64+d] --------
        {
            const int key = tid & 63;
            const int db  = tid >> 6;                      // 0..1
#pragma unroll
            for (int c = 0; c < 4; ++c) {
                const int d0 = db * 8 + c * 16;
                const __bf16* src = tile + key * 128 + 64 + d0;
#pragma unroll
                for (int j = 0; j < 8; ++j) VT[d0 + j][key] = src[j];
            }
        }
        __syncthreads();                                   // VT ready

        // ---- scores: four 16x16 tiles over 64 keys ------------------------
        v8f s[4];
#pragma unroll
        for (int kt = 0; kt < 4; ++kt) {
            const __bf16* kr = tile + (size_t)(kt * 16 + mn) * 128;
            v8f sv = vzero;
            sv = wmma_bf16(qa0, ldfrag(kr + half * 8,      kr + 16 + half * 8), sv);
            sv = wmma_bf16(qa1, ldfrag(kr + 32 + half * 8, kr + 48 + half * 8), sv);
            s[kt] = sv;
        }
        // ---- online softmax in exp2 domain --------------------------------
        float corr[8];
#pragma unroll
        for (int r = 0; r < 8; ++r) {
#pragma unroll
            for (int kt = 0; kt < 4; ++kt) s[kt][r] *= SC2;
            float v = fmaxf(fmaxf(s[0][r], s[1][r]), fmaxf(s[2][r], s[3][r]));
            v = fmaxf(v, __shfl_xor(v, 1, 32));
            v = fmaxf(v, __shfl_xor(v, 2, 32));
            v = fmaxf(v, __shfl_xor(v, 4, 32));
            v = fmaxf(v, __shfl_xor(v, 8, 32));
            const float mnew = fmaxf(mrow[r], v);
            corr[r] = exp2f(mrow[r] - mnew);
            mrow[r] = mnew;
            float rs = 0.f;
#pragma unroll
            for (int kt = 0; kt < 4; ++kt) {
                s[kt][r] = exp2f(s[kt][r] - mnew);
                rs += s[kt][r];
            }
            rs += __shfl_xor(rs, 1, 32);
            rs += __shfl_xor(rs, 2, 32);
            rs += __shfl_xor(rs, 4, 32);
            rs += __shfl_xor(rs, 8, 32);
            lrow[r] = lrow[r] * corr[r] + rs;
        }
#pragma unroll
        for (int t = 0; t < 4; ++t)
#pragma unroll
            for (int r = 0; r < 8; ++r) acc[t][r] *= corr[r];

        // ---- transpose P (D-layout -> A-layout 16x64) through LDS ---------
        __bf16 (*P)[64] = Plds[wave];
#pragma unroll
        for (int kt = 0; kt < 4; ++kt)
#pragma unroll
            for (int r = 0; r < 8; ++r)
                P[r + 8 * half][kt * 16 + mn] = f2bf(s[kt][r]);
        const v16bf pa0 = ldfrag(&P[mn][half * 8],      &P[mn][16 + half * 8]);
        const v16bf pa1 = ldfrag(&P[mn][32 + half * 8], &P[mn][48 + half * 8]);

        // ---- ctx += P(16x64) * V(64x64)  (V from VT, b128 loads) ----------
#pragma unroll
        for (int t = 0; t < 4; ++t) {
            const __bf16* vt = &VT[t * 16 + mn][0];
            acc[t] = wmma_bf16(pa0, ldfrag(vt + half * 8,      vt + 16 + half * 8), acc[t]);
            acc[t] = wmma_bf16(pa1, ldfrag(vt + 32 + half * 8, vt + 48 + half * 8), acc[t]);
        }
        __syncthreads();           // done with tile[cur] + VT before reuse
    }

    // ---- finalize: ctx[b, s, h*64 + d] as bf16 -----------------------------
#pragma unroll
    for (int r = 0; r < 8; ++r) {
        const float inv = 1.0f / lrow[r];
        const int row = qt * 16 + r + 8 * half;
        const size_t base = ((size_t)(b * SEQ + row)) * D_MODEL + h * D_K;
#pragma unroll
        for (int t = 0; t < 4; ++t)
            ctx[base + t * 16 + mn] = f2bfbits(acc[t][r] * inv);
    }
}

// ---------------------------------------------------------------------------
extern "C" void kernel_launch(void* const* d_in, const int* in_sizes, int n_in,
                              void* d_out, int out_size, void* d_ws, size_t ws_size,
                              hipStream_t stream) {
    const float* Q    = (const float*)d_in[0];
    const float* Kin  = (const float*)d_in[1];   // V input (d_in[2]) unused
    const float* W_q  = (const float*)d_in[3];
    const float* b_q  = (const float*)d_in[4];
    const float* W_kv = (const float*)d_in[5];
    const float* b_kv = (const float*)d_in[6];
    const float* W_o  = (const float*)d_in[7];
    const float* b_o  = (const float*)d_in[8];

    unsigned short* p    = (unsigned short*)d_ws;
    unsigned short* Qb   = p; p += (size_t)ROWS * D_MODEL;
    unsigned short* Kb   = p; p += (size_t)ROWS * D_MODEL;
    unsigned short* Wqb  = p; p += (size_t)D_MODEL * D_MODEL;
    unsigned short* Wkvb = p; p += (size_t)128 * D_MODEL;
    unsigned short* Wob  = p; p += (size_t)D_MODEL * D_MODEL;
    unsigned short* qws  = p; p += (size_t)ROWS * D_MODEL;
    unsigned short* kvws = p; p += (size_t)ROWS * 128;
    unsigned short* ctx  = p;                       // ROWS x D_MODEL bf16

    auto cvt = [&](const float* s, unsigned short* d, size_t n) {
        const int n8 = (int)(n / 8);
        cvt_bf16<<<dim3((n8 + 255) / 256), 256, 0, stream>>>(s, d, n8);
    };
    cvt(Q,    Qb,   (size_t)ROWS * D_MODEL);
    cvt(Kin,  Kb,   (size_t)ROWS * D_MODEL);
    cvt(W_q,  Wqb,  (size_t)D_MODEL * D_MODEL);
    cvt(W_kv, Wkvb, (size_t)128 * D_MODEL);
    cvt(W_o,  Wob,  (size_t)D_MODEL * D_MODEL);

    gemm_bf16_wmma<true><<<dim3(D_MODEL / 128, ROWS / 64), 128, 0, stream>>>(
        Qb, Wqb, b_q, qws, ROWS, D_MODEL, D_MODEL);
    gemm_bf16_wmma<true><<<dim3(1, ROWS / 64), 128, 0, stream>>>(
        Kb, Wkvb, b_kv, kvws, ROWS, 128, D_MODEL);
    attn_kernel<<<dim3(1024), 128, 0, stream>>>(qws, kvws, ctx);
    gemm_bf16_wmma<false><<<dim3(D_MODEL / 128, ROWS / 64), 128, 0, stream>>>(
        ctx, Wob, b_o, d_out, ROWS, D_MODEL, D_MODEL);
}
